// SoccerNetTransformerCaption_90683939488529
// MI455X (gfx1250) — matrix operand order repaired and checked by reading, not compiled, not run on hardware
//
#include <hip/hip_runtime.h>
#include <hip/hip_bf16.h>

// MI455X / gfx1250 wave32 WMMA implementation of the LSTM caption decoder.
// All GEMMs use v_wmma_f32_16x16x32_bf16 on pre-swizzled bf16 fragments.

typedef __attribute__((ext_vector_type(16))) __bf16 v16bf;
typedef __attribute__((ext_vector_type(8)))  float  v8f;

#define DEV __device__ __forceinline__

// ---- WMMA fragment index helpers (ISA 7.12.2, wave32) -----------------
// A matrix (MxK), 16x32 tiles. Lane = (m&15) + 16*g ; element j in v16bf.
// k-in-tile mapping: g=0: j0..7 -> k0..7, j8..15 -> k16..23
//                    g=1: j0..7 -> k8..15, j8..15 -> k24..31
DEV size_t afrag_idx(int m, int k, int KT) {
  int mt = m >> 4, kt = k >> 5, ko = k & 31;
  int g, j;
  if (ko < 8)       { g = 0; j = ko;      }
  else if (ko < 16) { g = 1; j = ko - 8;  }
  else if (ko < 24) { g = 0; j = ko - 8;  }
  else              { g = 1; j = ko - 16; }
  int lane = (m & 15) + (g << 4);
  return (((size_t)mt * KT + kt) * 32 + lane) * 16 + j;
}
// B matrix (KxN), 32x16 tiles. Lane = (n&15) + 16*((ko>>3)&1); element j = (ko&7) + 8*(ko>=16).
DEV size_t bfrag_idx(int k, int n, int KT) {
  int kt = k >> 5, ko = k & 31, nt = n >> 4;
  int lane = (n & 15) + (((ko >> 3) & 1) << 4);
  return (((size_t)nt * KT + kt) * 32 + lane) * 16 + ((ko & 7) + ((ko >> 4) << 3));
}

DEV float sigmoidf_(float x) { return 1.0f / (1.0f + __expf(-x)); }

// ---- conversion kernels ----------------------------------------------
// w is (N,K) row-major fp32; build B-fragments for B = w^T (KxN), N padded.
__global__ void wT_to_bfrag(const float* __restrict__ w, int N, int K,
                            __bf16* __restrict__ dst, int NT_pad) {
  size_t idx = (size_t)blockIdx.x * 256 + threadIdx.x;
  size_t total = (size_t)NT_pad * 16 * K;
  if (idx >= total) return;
  int n = (int)(idx / K);
  int k = (int)(idx % K);
  float v = (n < N) ? w[(size_t)n * K + k] : 0.0f;
  dst[bfrag_idx(k, n, K >> 5)] = (__bf16)v;
}

// src is (M,K) row-major fp32 -> A fragments.
__global__ void f32_to_afrag(const float* __restrict__ src, int M, int K,
                             __bf16* __restrict__ dst) {
  size_t idx = (size_t)blockIdx.x * 256 + threadIdx.x;
  if (idx >= (size_t)M * K) return;
  int m = (int)(idx / K);
  int k = (int)(idx % K);
  dst[afrag_idx(m, k, K >> 5)] = (__bf16)src[idx];
}

// ---- generic dual-input WMMA GEMM: out = A0@B0 + A1@B1 + bias (+relu) --
// grid = (N/256, M/32), 256 threads (8 waves: 2 m-waves x 4 n-waves).
__global__ __launch_bounds__(256) void gemm_wmma(
    const __bf16* __restrict__ A0, const __bf16* __restrict__ B0, int KT0,
    const __bf16* __restrict__ A1, const __bf16* __restrict__ B1, int KT1,
    const float* __restrict__ bias0, const float* __restrict__ bias1,
    float* __restrict__ out, int N, int relu) {
  int lane = threadIdx.x & 31;
  int w    = threadIdx.x >> 5;
  int wm = w >> 2, wn = w & 3;
  int mt  = blockIdx.y * 2 + wm;
  int nt0 = blockIdx.x * 16 + wn * 4;

  v8f acc[4] = {};
  for (int kt = 0; kt < KT0; ++kt) {
    v16bf a = *(const v16bf*)(A0 + (((size_t)mt * KT0 + kt) * 32 + lane) * 16);
#pragma unroll
    for (int i = 0; i < 4; ++i) {
      v16bf b = *(const v16bf*)(B0 + (((size_t)(nt0 + i) * KT0 + kt) * 32 + lane) * 16);
      acc[i] = __builtin_amdgcn_wmma_f32_16x16x32_bf16(false, a, false, b,
                                                       (short)0, acc[i], false, false);
    }
  }
  for (int kt = 0; kt < KT1; ++kt) {
    v16bf a = *(const v16bf*)(A1 + (((size_t)mt * KT1 + kt) * 32 + lane) * 16);
#pragma unroll
    for (int i = 0; i < 4; ++i) {
      v16bf b = *(const v16bf*)(B1 + (((size_t)(nt0 + i) * KT1 + kt) * 32 + lane) * 16);
      acc[i] = __builtin_amdgcn_wmma_f32_16x16x32_bf16(false, a, false, b,
                                                       (short)0, acc[i], false, false);
    }
  }
  int col = lane & 15, g = lane >> 4;
#pragma unroll
  for (int i = 0; i < 4; ++i) {
    int n = (nt0 + i) * 16 + col;
    float bs = bias0 ? bias0[n] : 0.0f;
    if (bias1) bs += bias1[n];
#pragma unroll
    for (int r = 0; r < 8; ++r) {
      int m = mt * 16 + r + 8 * g;
      float v = acc[i][r] + bs;
      if (relu) v = fmaxf(v, 0.0f);
      out[(size_t)m * N + n] = v;
    }
  }
}

// ---- final vocab projection: (2048 x 512) @ (512 x 20000) + mask -------
// grid = (79, 64), 256 threads. N padded to 1264 tiles (20224 cols).
__global__ __launch_bounds__(256) void gemm_fc(
    const __bf16* __restrict__ A, const __bf16* __restrict__ B,
    const float* __restrict__ bias, const int* __restrict__ lengths,
    float* __restrict__ out) {
  const int KT = 16, V = 20000;
  int lane = threadIdx.x & 31;
  int w    = threadIdx.x >> 5;
  int wm = w >> 2, wn = w & 3;
  int mt  = blockIdx.y * 2 + wm;
  int nt0 = blockIdx.x * 16 + wn * 4;

  v8f acc[4] = {};
  for (int kt = 0; kt < KT; ++kt) {
    v16bf a = *(const v16bf*)(A + (((size_t)mt * KT + kt) * 32 + lane) * 16);
#pragma unroll
    for (int i = 0; i < 4; ++i) {
      v16bf b = *(const v16bf*)(B + (((size_t)(nt0 + i) * KT + kt) * 32 + lane) * 16);
      acc[i] = __builtin_amdgcn_wmma_f32_16x16x32_bf16(false, a, false, b,
                                                       (short)0, acc[i], false, false);
    }
  }
  int col = lane & 15, g = lane >> 4;
#pragma unroll
  for (int i = 0; i < 4; ++i) {
    int n = (nt0 + i) * 16 + col;
    if (n >= V) continue;
    float bs = bias[n];
#pragma unroll
    for (int r = 0; r < 8; ++r) {
      int m = mt * 16 + r + 8 * g;          // m = b*32 + l
      int bb = m >> 5, ll = m & 31;
      bool on = ll < lengths[bb];
      out[(size_t)m * V + n] = on ? (acc[i][r] + bs) : 0.0f;
    }
  }
}

// ---- replicate h_init into both layers' h/c and their A-fragments ------
__global__ void init_state(const float* __restrict__ hi,
                           float* h0, float* c0, float* h1, float* c1,
                           __bf16* h0f, __bf16* h1f) {
  int idx = blockIdx.x * 256 + threadIdx.x;   // 64*512 total
  float v = hi[idx];
  h0[idx] = v; c0[idx] = v; h1[idx] = v; c1[idx] = v;
  int m = idx >> 9, k = idx & 511;
  __bf16 bv = (__bf16)v;
  size_t fi = afrag_idx(m, k, 16);
  h0f[fi] = bv; h1f[fi] = bv;
}

// ---- per-step attention + softmax + context + build inp fragments ------
__global__ __launch_bounds__(256) void attn_concat(
    const float* __restrict__ enc, const float* __restrict__ h1,
    const int* __restrict__ captions, const float* __restrict__ embed,
    int l, __bf16* __restrict__ inpf) {
  int b = blockIdx.x, tid = threadIdx.x;
  __shared__ float sh[512];
  __shared__ float sw[512];
  __shared__ float red[256];
  sh[tid]       = h1[b * 512 + tid];
  sh[tid + 256] = h1[b * 512 + tid + 256];
  __syncthreads();

  const float scale = 0.04419417382415922f;   // 1/sqrt(512)
  const float* encb = enc + (size_t)b * 512 * 512;
  for (int t = tid; t < 512; t += 256) {
    const float* er = encb + (size_t)t * 512;
    float acc = 0.0f;
    for (int h = 0; h < 512; h += 4)
      acc += sh[h] * er[h] + sh[h+1] * er[h+1] + sh[h+2] * er[h+2] + sh[h+3] * er[h+3];
    sw[t] = acc * scale;
  }
  __syncthreads();
  // softmax over 512
  red[tid] = fmaxf(sw[tid], sw[tid + 256]);
  __syncthreads();
  for (int s = 128; s > 0; s >>= 1) { if (tid < s) red[tid] = fmaxf(red[tid], red[tid+s]); __syncthreads(); }
  float mx = red[0];
  __syncthreads();
  float e0 = __expf(sw[tid] - mx), e1 = __expf(sw[tid + 256] - mx);
  red[tid] = e0 + e1;
  __syncthreads();
  for (int s = 128; s > 0; s >>= 1) { if (tid < s) red[tid] += red[tid+s]; __syncthreads(); }
  float inv = 1.0f / red[0];
  __syncthreads();
  sw[tid] = e0 * inv; sw[tid + 256] = e1 * inv;
  __syncthreads();
  // ctx = sum_t w[t] * enc[b,t,:]
  float a0 = 0.0f, a1 = 0.0f;
  for (int t = 0; t < 512; ++t) {
    float wt = sw[t];
    const float* er = encb + (size_t)t * 512;
    a0 += wt * er[tid];
    a1 += wt * er[tid + 256];
  }
  // inp = [embed(word) | ctx] row b, K=1024 (KT=32), written as A-fragments
  int wid = captions[b * 32 + l];
  const float* wr = embed + (size_t)wid * 512;
  inpf[afrag_idx(b, tid,        32)] = (__bf16)wr[tid];
  inpf[afrag_idx(b, tid + 256,  32)] = (__bf16)wr[tid + 256];
  inpf[afrag_idx(b, 512 + tid,       32)] = (__bf16)a0;
  inpf[afrag_idx(b, 512 + tid + 256, 32)] = (__bf16)a1;
}

// ---- LSTM cell elementwise (PyTorch gate order i,f,g,o) ----------------
__global__ void lstm_cell(const float* __restrict__ gates,
                          float* __restrict__ c, float* __restrict__ h,
                          __bf16* __restrict__ hf,
                          __bf16* __restrict__ hidf, int l, int writeHid) {
  int b = blockIdx.x, tid = threadIdx.x;
  const float* g = gates + (size_t)b * 2048;
#pragma unroll
  for (int q = 0; q < 2; ++q) {
    int j = tid + q * 256;
    float ig = g[j], fg = g[512 + j], gg = g[1024 + j], og = g[1536 + j];
    float cc = c[b * 512 + j];
    float c2 = sigmoidf_(fg) * cc + sigmoidf_(ig) * tanhf(gg);
    float h2 = sigmoidf_(og) * tanhf(c2);
    c[b * 512 + j] = c2;
    h[b * 512 + j] = h2;
    __bf16 hb = (__bf16)h2;
    hf[afrag_idx(b, j, 16)] = hb;
    if (writeHid) hidf[afrag_idx(b * 32 + l, j, 16)] = hb;
  }
}

extern "C" void kernel_launch(void* const* d_in, const int* in_sizes, int n_in,
                              void* d_out, int out_size, void* d_ws, size_t ws_size,
                              hipStream_t stream) {
  const float* features = (const float*)d_in[0];
  const int*   captions = (const int*)  d_in[1];
  const int*   lengths  = (const int*)  d_in[2];
  const float* enc      = (const float*)d_in[3];
  const float* embed    = (const float*)d_in[4];
  const float* ft1_w = (const float*)d_in[5];  const float* ft1_b = (const float*)d_in[6];
  const float* ft2_w = (const float*)d_in[7];  const float* ft2_b = (const float*)d_in[8];
  const float* w_ih0 = (const float*)d_in[9];  const float* w_hh0 = (const float*)d_in[10];
  const float* b_ih0 = (const float*)d_in[11]; const float* b_hh0 = (const float*)d_in[12];
  const float* w_ih1 = (const float*)d_in[13]; const float* w_hh1 = (const float*)d_in[14];
  const float* b_ih1 = (const float*)d_in[15]; const float* b_hh1 = (const float*)d_in[16];
  const float* fc_w  = (const float*)d_in[17]; const float* fc_b  = (const float*)d_in[18];
  float* out = (float*)d_out;

  // ---- workspace carve-up (~37 MB, 256B aligned) ----
  char* ws = (char*)d_ws; size_t off = 0;
  auto alloc = [&](size_t bytes) -> char* {
    char* p = ws + off; off = (off + bytes + 255) & ~(size_t)255; return p;
  };
  __bf16* Wft1T = (__bf16*)alloc((size_t)32  * 16 * 512 * 2);
  __bf16* Wft2T = (__bf16*)alloc((size_t)32  * 16 * 512 * 2);
  __bf16* Wih0T = (__bf16*)alloc((size_t)128 * 32 * 512 * 2);
  __bf16* Whh0T = (__bf16*)alloc((size_t)128 * 16 * 512 * 2);
  __bf16* Wih1T = (__bf16*)alloc((size_t)128 * 16 * 512 * 2);
  __bf16* Whh1T = (__bf16*)alloc((size_t)128 * 16 * 512 * 2);
  __bf16* fcT   = (__bf16*)alloc((size_t)1264 * 16 * 512 * 2);
  __bf16* featF = (__bf16*)alloc((size_t)4 * 16 * 512 * 2);
  __bf16* xF    = (__bf16*)alloc((size_t)4 * 16 * 512 * 2);
  __bf16* inpF  = (__bf16*)alloc((size_t)4 * 32 * 512 * 2);
  __bf16* h0F   = (__bf16*)alloc((size_t)4 * 16 * 512 * 2);
  __bf16* h1F   = (__bf16*)alloc((size_t)4 * 16 * 512 * 2);
  __bf16* hidF  = (__bf16*)alloc((size_t)128 * 16 * 512 * 2);
  float* xbuf  = (float*)alloc((size_t)64 * 512 * 4);
  float* hinit = (float*)alloc((size_t)64 * 512 * 4);
  float* h0 = (float*)alloc((size_t)64 * 512 * 4);
  float* c0 = (float*)alloc((size_t)64 * 512 * 4);
  float* h1 = (float*)alloc((size_t)64 * 512 * 4);
  float* c1 = (float*)alloc((size_t)64 * 512 * 4);
  float* gates = (float*)alloc((size_t)64 * 2048 * 4);

  dim3 blk(256);
  auto nb = [](size_t total) { return (unsigned)((total + 255) / 256); };

  // ---- weight conversion to bf16 B-fragments ----
  wT_to_bfrag<<<nb((size_t)32  * 16 * 512),  blk, 0, stream>>>(ft1_w, 512,  512,  Wft1T, 32);
  wT_to_bfrag<<<nb((size_t)32  * 16 * 512),  blk, 0, stream>>>(ft2_w, 512,  512,  Wft2T, 32);
  wT_to_bfrag<<<nb((size_t)128 * 16 * 1024), blk, 0, stream>>>(w_ih0, 2048, 1024, Wih0T, 128);
  wT_to_bfrag<<<nb((size_t)128 * 16 * 512),  blk, 0, stream>>>(w_hh0, 2048, 512,  Whh0T, 128);
  wT_to_bfrag<<<nb((size_t)128 * 16 * 512),  blk, 0, stream>>>(w_ih1, 2048, 512,  Wih1T, 128);
  wT_to_bfrag<<<nb((size_t)128 * 16 * 512),  blk, 0, stream>>>(w_hh1, 2048, 512,  Whh1T, 128);
  wT_to_bfrag<<<nb((size_t)1264 * 16 * 512), blk, 0, stream>>>(fc_w, 20000, 512,  fcT,  1264);

  // ---- feature MLP -> h_init ----
  f32_to_afrag<<<nb((size_t)64 * 512), blk, 0, stream>>>(features, 64, 512, featF);
  gemm_wmma<<<dim3(2, 2), blk, 0, stream>>>(featF, Wft1T, 16, nullptr, nullptr, 0,
                                            ft1_b, nullptr, xbuf, 512, 1);
  f32_to_afrag<<<nb((size_t)64 * 512), blk, 0, stream>>>(xbuf, 64, 512, xF);
  gemm_wmma<<<dim3(2, 2), blk, 0, stream>>>(xF, Wft2T, 16, nullptr, nullptr, 0,
                                            ft2_b, nullptr, hinit, 512, 0);
  init_state<<<128, blk, 0, stream>>>(hinit, h0, c0, h1, c1, h0F, h1F);

  // ---- 32 sequential decode steps ----
  for (int l = 0; l < 32; ++l) {
    attn_concat<<<64, blk, 0, stream>>>(enc, h1, captions, embed, l, inpF);
    gemm_wmma<<<dim3(8, 2), blk, 0, stream>>>(inpF, Wih0T, 32, h0F, Whh0T, 16,
                                              b_ih0, b_hh0, gates, 2048, 0);
    lstm_cell<<<64, blk, 0, stream>>>(gates, c0, h0, h0F, nullptr, 0, 0);
    gemm_wmma<<<dim3(8, 2), blk, 0, stream>>>(h0F, Wih1T, 16, h1F, Whh1T, 16,
                                              b_ih1, b_hh1, gates, 2048, 0);
    lstm_cell<<<64, blk, 0, stream>>>(gates, c1, h1, h1F, hidF, l, 1);
  }

  // ---- vocab projection with length mask (writes every output element) ----
  gemm_fc<<<dim3(79, 64), blk, 0, stream>>>(hidF, fcT, fc_b, lengths, out);
}
